// NeuralNet_48249662603615
// MI455X (gfx1250) — matrix operand order repaired and checked by reading, not compiled, not run on hardware
//
#include <hip/hip_runtime.h>

typedef __attribute__((ext_vector_type(16))) _Float16 v16h;
typedef __attribute__((ext_vector_type(8)))  float    v8f;

#define WPB 4            // waves per block
#define THREADS (WPB*32)
#define FDIM 128
#define MMEM 16
#define MS 132           // padded LDS row stride (floats): 16 rows cover all 64 banks

__global__ __launch_bounds__(THREADS)
void agree_group_fwd(const int* __restrict__ groups, const int* __restrict__ movies,
                     const int* __restrict__ members,
                     const float* __restrict__ user_emb, const float* __restrict__ movie_emb,
                     const float* __restrict__ group_emb,
                     const float* __restrict__ att_w1, const float* __restrict__ att_b1,
                     const float* __restrict__ att_w2, const float* __restrict__ att_b2,
                     const float* __restrict__ pred_w1, const float* __restrict__ pred_b1,
                     const float* __restrict__ pred_w2, const float* __restrict__ pred_b2,
                     float* __restrict__ out, int B)
{
    // att_w1 pre-packed into WMMA B-fragment layout:
    // attP[(cb*32 + lane)*16 + i] = f16(att_w1[(cb*32 + 8*(lane>>4) + i + (i>=8 ? 8 : 0))*16 + (lane&15)])
    // -> per chunk, B operand is ONE aligned 32B v16h LDS load.
    __shared__ __attribute__((aligned(32))) _Float16 attP[8 * 32 * 16];   // 8 KB
    __shared__ float predT[8 * 384];                                      // pred_w1 transposed (12 KB)
    __shared__ __attribute__((aligned(16))) float memS[WPB][MMEM * MS];   // per-wave member tile (f32, reused twice)
    __shared__ __attribute__((aligned(16))) float itemS[WPB][FDIM];       // per-wave item row

    const int tid  = threadIdx.x;
    const int lane = tid & 31;
    const int wv   = tid >> 5;
    const int hi   = lane >> 4;     // half-wave select
    const int nl   = lane & 15;     // A row m / B,C column n

    // ---- block-shared weight staging (L2-hot, amortized over 4 waves) ----
    // pack att_w1 into B-fragment layout: 256 (chunk,lane) pairs of 16 f16
    for (int p = tid; p < 8 * 32; p += THREADS) {
        const int cb2 = p >> 5;
        const int l2  = p & 31;
        const int krow = cb2 * 32 + 8 * (l2 >> 4);
        const int n2   = l2 & 15;
        #pragma unroll
        for (int i = 0; i < 16; ++i)
            attP[p * 16 + i] = (_Float16)att_w1[(krow + i + (i >= 8 ? 8 : 0)) * 16 + n2];
    }
    // transpose pred_w1 -> predT[j][f] (kills 16-way bank conflicts later)
    for (int i = tid; i < 8 * 384; i += THREADS) {
        const int j = i / 384;
        const int f = i - j * 384;
        predT[i] = pred_w1[f * 8 + j];
    }

    int b = blockIdx.x * WPB + wv;
    if (b >= B) b = B - 1;          // clamp: duplicate waves write identical values
    const int gid = groups[b];
    const int mv  = movies[b];

    // ---- stage item row (keep in regs too) ----
    const float4 it = *(const float4*)(movie_emb + (size_t)mv * FDIM + lane * 4);
    *(float4*)&itemS[wv][lane * 4] = it;

    // ---- stage 16x128 member tile: row t, float4 chunk = lane ----
    #pragma unroll
    for (int t = 0; t < MMEM; ++t) {
        const int uid = members[gid * MMEM + t];
        float4 v = *(const float4*)(user_emb + (size_t)uid * FDIM + lane * 4);
        *(float4*)&memS[wv][t * MS + lane * 4] = v;
    }
    __syncthreads();   // uniform across all waves; makes LDS writes cross-lane visible

    // ---- attention GEMM: [16x256] x [256x16] via 8x v_wmma_f32_16x16x32_f16 ----
    v8f acc = {0.f, 0.f, 0.f, 0.f, 0.f, 0.f, 0.f, 0.f};
    const float* arow = &memS[wv][nl * MS];
    #pragma unroll
    for (int cb = 0; cb < 8; ++cb) {
        const int kbase = cb * 32;
        // A rows for K>=128 are the (broadcast) item row
        const float* src = (kbase < 128) ? (arow + kbase) : (&itemS[wv][kbase - 128]);
        const int o1 = hi * 8;
        float4 q0 = *(const float4*)(src + o1);
        float4 q1 = *(const float4*)(src + o1 + 4);
        float4 q2 = *(const float4*)(src + 16 + o1);
        float4 q3 = *(const float4*)(src + 16 + o1 + 4);
        v16h a;
        a[0]=(_Float16)q0.x; a[1]=(_Float16)q0.y; a[2]=(_Float16)q0.z; a[3]=(_Float16)q0.w;
        a[4]=(_Float16)q1.x; a[5]=(_Float16)q1.y; a[6]=(_Float16)q1.z; a[7]=(_Float16)q1.w;
        a[8]=(_Float16)q2.x; a[9]=(_Float16)q2.y; a[10]=(_Float16)q2.z; a[11]=(_Float16)q2.w;
        a[12]=(_Float16)q3.x; a[13]=(_Float16)q3.y; a[14]=(_Float16)q3.z; a[15]=(_Float16)q3.w;
        const v16h bb = *(const v16h*)&attP[(cb * 32 + lane) * 16];
        acc = __builtin_amdgcn_wmma_f32_16x16x32_f16(false, a, false, bb,
                                                     (short)0, acc, false, false);
    }

    // ---- bias + relu + dot(att_w2) per column n, reduce over the 16 columns ----
    const float b1n = att_b1[nl];
    const float w2n = att_w2[nl];
    float sc[8];
    #pragma unroll
    for (int r = 0; r < 8; ++r) {
        float h = acc[r] + b1n;
        h = h > 0.f ? h : 0.f;
        float p = h * w2n;
        p += __shfl_xor(p, 1);
        p += __shfl_xor(p, 2);
        p += __shfl_xor(p, 4);
        p += __shfl_xor(p, 8);
        sc[r] = p;                 // score for member m = r + 8*hi (replicated in half-wave)
    }

    // ---- softmax over the 16 members (8 local + cross-half shuffle) ----
    float mx = sc[0];
    #pragma unroll
    for (int r = 1; r < 8; ++r) mx = fmaxf(mx, sc[r]);
    mx = fmaxf(mx, __shfl_xor(mx, 16));
    float e[8], ssum = 0.f;
    #pragma unroll
    for (int r = 0; r < 8; ++r) { e[r] = __expf(sc[r] - mx); ssum += e[r]; }
    ssum += __shfl_xor(ssum, 16);
    const float inv = 1.0f / ssum;
    float wtA[8], wtB[8];
    #pragma unroll
    for (int r = 0; r < 8; ++r) {
        float w = e[r] * inv;
        wtA[r] = __shfl(w, 0);     // wt[m=r]    (from lower half)
        wtB[r] = __shfl(w, 16);    // wt[m=8+r]  (from upper half)
    }

    // ---- weighted member sum (reuse LDS tile) + group embedding ----
    const int f0 = lane * 4;
    float4 ge = *(const float4*)(group_emb + (size_t)gid * FDIM + f0);
    float gacc[4] = { ge.x, ge.y, ge.z, ge.w };
    #pragma unroll
    for (int r = 0; r < 8; ++r) {
        float4 m0 = *(const float4*)&memS[wv][r * MS + f0];
        float4 m1 = *(const float4*)&memS[wv][(8 + r) * MS + f0];
        gacc[0] += wtA[r] * m0.x + wtB[r] * m1.x;
        gacc[1] += wtA[r] * m0.y + wtB[r] * m1.y;
        gacc[2] += wtA[r] * m0.z + wtB[r] * m1.z;
        gacc[3] += wtA[r] * m0.w + wtB[r] * m1.w;
    }

    // ---- prediction MLP: feat = [g*item, g, item] (384) -> 8 -> 1 -> sigmoid ----
    const float itv[4] = { it.x, it.y, it.z, it.w };
    float h2[8] = {0.f,0.f,0.f,0.f,0.f,0.f,0.f,0.f};
    #pragma unroll
    for (int j = 0; j < 8; ++j) {
        const float* wrow = &predT[j * 384];
        float s = 0.f;
        #pragma unroll
        for (int i = 0; i < 4; ++i) {
            const int f = f0 + i;
            s += (gacc[i] * itv[i]) * wrow[f]
               + gacc[i] * wrow[128 + f]
               + itv[i]  * wrow[256 + f];
        }
        h2[j] = s;
    }
    #pragma unroll
    for (int j = 0; j < 8; ++j) {
        h2[j] += __shfl_xor(h2[j], 1);
        h2[j] += __shfl_xor(h2[j], 2);
        h2[j] += __shfl_xor(h2[j], 4);
        h2[j] += __shfl_xor(h2[j], 8);
        h2[j] += __shfl_xor(h2[j], 16);
    }
    float o = pred_b2[0];
    #pragma unroll
    for (int j = 0; j < 8; ++j) {
        float v = h2[j] + pred_b1[j];
        v = v > 0.f ? v : 0.f;
        o += v * pred_w2[j];
    }
    o = 1.0f / (1.0f + __expf(-o));
    if (lane == 0) out[b] = o;
}

extern "C" void kernel_launch(void* const* d_in, const int* in_sizes, int n_in,
                              void* d_out, int out_size, void* d_ws, size_t ws_size,
                              hipStream_t stream) {
    const int*   groups    = (const int*)d_in[0];
    const int*   movies    = (const int*)d_in[1];
    const int*   members   = (const int*)d_in[2];
    const float* user_emb  = (const float*)d_in[3];
    const float* movie_emb = (const float*)d_in[4];
    const float* group_emb = (const float*)d_in[5];
    const float* att_w1    = (const float*)d_in[6];
    const float* att_b1    = (const float*)d_in[7];
    const float* att_w2    = (const float*)d_in[8];
    const float* att_b2    = (const float*)d_in[9];
    const float* pred_w1   = (const float*)d_in[10];
    const float* pred_b1   = (const float*)d_in[11];
    const float* pred_w2   = (const float*)d_in[12];
    const float* pred_b2   = (const float*)d_in[13];
    float* out = (float*)d_out;
    const int B = in_sizes[0];
    const int grid = (B + WPB - 1) / WPB;
    agree_group_fwd<<<grid, THREADS, 0, stream>>>(
        groups, movies, members, user_emb, movie_emb, group_emb,
        att_w1, att_b1, att_w2, att_b2,
        pred_w1, pred_b1, pred_w2, pred_b2, out, B);
    (void)att_b2; (void)d_ws; (void)ws_size; (void)n_in; (void)out_size;
}
